// RGCNLayer_66778151518270
// MI455X (gfx1250) — matrix operand order repaired and checked by reading, not compiled, not run on hardware
//
#include <hip/hip_runtime.h>

#define NN 50000
#define EE 800000
#define DD 128
#define RR 8
#define MTILES ((NN + 15) / 16)   // 3125 exactly

typedef __attribute__((ext_vector_type(16))) __bf16 v16bf;
typedef __attribute__((ext_vector_type(8)))  float  v8f;

__device__ __forceinline__ unsigned short f2bf(float f) {
  union { float f; unsigned u; } c; c.f = f;
  unsigned u = c.u;
  unsigned r = u + 0x7FFFu + ((u >> 16) & 1u);   // round-to-nearest-even
  return (unsigned short)(r >> 16);
}

union Frag { v16bf v; uint4 q[2]; };

// ---------------------------------------------------------------------------
// Prep 1: x (f32) -> xb (bf16, row-major [N][128]). Pure streaming.
// ---------------------------------------------------------------------------
__global__ void rgcn_cvt_x(const float* __restrict__ x,
                           unsigned short* __restrict__ xb) {
  const int i = blockIdx.x * 256 + threadIdx.x;   // one float4 / 4 bf16 each
  if (i >= NN * DD / 4) return;
  const float4 p = ((const float4*)x)[i];
  ushort4 o;
  o.x = f2bf(p.x); o.y = f2bf(p.y); o.z = f2bf(p.z); o.w = f2bf(p.w);
  ((ushort4*)xb)[i] = o;
}

// ---------------------------------------------------------------------------
// Prep 2: weights -> bf16, n-major: wtb[mat][n][k]
//   mat 0..7 : wtb[mat][n][k] = Wrel[mat][k][n]   (transpose)
//   mat 8    : wtb[8][n][k]   = Wself[n][k]       (already n-major)
// 9*128*128*2 B = 295 KB -> lives in L2 for the GEMM kernel.
// ---------------------------------------------------------------------------
__global__ void rgcn_cvt_w(const float* __restrict__ Wrel,
                           const float* __restrict__ Wself,
                           unsigned short* __restrict__ wtb) {
  const int i = blockIdx.x * 256 + threadIdx.x;   // element index
  if (i >= (RR + 1) * DD * DD) return;
  const int mat = i >> 14;          // /16384
  const int n   = (i >> 7) & 127;
  const int k   = i & 127;
  float v;
  if (mat < RR) v = Wrel[(size_t)mat * DD * DD + (size_t)k * DD + n];
  else          v = Wself[(size_t)n * DD + k];
  wtb[i] = f2bf(v);
}

// ---------------------------------------------------------------------------
// Kernel 1: 9 GEMMs via v_wmma_f32_16x16x32_bf16.
//   blockIdx.y = mat in [0,8]; mat<8 -> h[mat] = x @ W_rel[mat] (to ws),
//   mat==8    -> d_out = x @ W_self^T + b (pre-activation staging).
// One wave per 16-row tile, full 16x128 output stripe.
// All fragments are two global_load_b128s; zero per-element VALU marshalling.
// ---------------------------------------------------------------------------
__global__ void rgcn_gemm_wmma(const unsigned short* __restrict__ xb,
                               const unsigned short* __restrict__ wtb,
                               const float* __restrict__ bself,
                               float* __restrict__ hbuf,
                               float* __restrict__ outbuf) {
  const int lane  = threadIdx.x & 31;
  const int wave  = threadIdx.x >> 5;
  const int mtile = blockIdx.x * 4 + wave;
  if (mtile >= MTILES) return;                 // wave-uniform: EXEC stays all-1s
  const int  mat  = blockIdx.y;                // 0..7 relation, 8 = self-loop
  const bool self = (mat == RR);
  const int  row0 = mtile * 16;

  v8f acc[8];
#pragma unroll
  for (int nt = 0; nt < 8; ++nt) acc[nt] = (v8f){0,0,0,0,0,0,0,0};

  const int n   = lane & 15;                   // also m for A side
  const int khA = (lane >> 4) << 3;            // A: +8  K for lanes 16-31
  const int khB = (lane >> 4) << 3;            // B: +8  K for lanes 16-31

  // A row base (bf16 elements): xb[(row0+m)*128 + ...]
  const unsigned short* arow = xb + (size_t)(row0 + n) * DD;
  // B matrix base (bf16 elements): wtb[mat][col][k]
  const unsigned short* wmat = wtb + (size_t)mat * DD * DD;

#pragma unroll
  for (int kc = 0; kc < 4; ++kc) {
    // A fragment: k = khA+0..7 in VGPRs 0-3, k = khA+16..23 in VGPRs 4-7
    Frag a;
    const unsigned short* ap = arow + kc * 32 + khA;
    a.q[0] = *(const uint4*)(ap);        // 8 bf16, 16B aligned
    a.q[1] = *(const uint4*)(ap + 16);
#pragma unroll
    for (int nt = 0; nt < 8; ++nt) {
      const int col = nt * 16 + n;
      // B fragment: k = khB+0..7 and khB+16..23 at column col
      Frag b;
      const unsigned short* bp = wmat + (size_t)col * DD + kc * 32 + khB;
      b.q[0] = *(const uint4*)(bp);
      b.q[1] = *(const uint4*)(bp + 16);
      acc[nt] = __builtin_amdgcn_wmma_f32_16x16x32_bf16(
          false, a.v, false, b.v, (short)0, acc[nt], false, false);
    }
  }

  // store: C/D layout -> VGPR j holds M=j (lanes 0-15) / M=j+8 (16-31)
  const int mo = (lane >> 4) << 3;
#pragma unroll
  for (int nt = 0; nt < 8; ++nt) {
    const int col  = nt * 16 + n;
    const float bd = self ? bself[col] : 0.0f;
#pragma unroll
    for (int j = 0; j < 8; ++j) {
      const int row = row0 + mo + j;
      const float vv = acc[nt][j] + bd;
      if (self) outbuf[(size_t)row * DD + col] = vv;
      else      hbuf[((size_t)mat * NN + row) * DD + col] = vv;
    }
  }
}

// ---------------------------------------------------------------------------
// Kernel 2: per-edge gather + scatter-add. One wave per edge:
// lane loads float4 of h[etype][src] (coalesced 512B), 4x global_atomic_add_f32
// into agg[dst] (25.6 MB, L2-resident). Lane 0 bumps degree.
// ---------------------------------------------------------------------------
__global__ void rgcn_edges(const int* __restrict__ eidx,
                           const int* __restrict__ etype,
                           const float* __restrict__ hbuf,
                           float* __restrict__ agg,
                           float* __restrict__ deg) {
  const int lane = threadIdx.x & 31;
  const long long e = (long long)blockIdx.x * 8 + (threadIdx.x >> 5);
  if (e >= EE) return;                         // wave-uniform
  const int s = eidx[e];
  const int d = eidx[(size_t)EE + e];
  const int r = etype[e];
  const float4 mv = ((const float4*)(hbuf + ((size_t)r * NN + s) * DD))[lane];
  float* ap = agg + (size_t)d * DD + lane * 4;
  unsafeAtomicAdd(ap + 0, mv.x);
  unsafeAtomicAdd(ap + 1, mv.y);
  unsafeAtomicAdd(ap + 2, mv.z);
  unsafeAtomicAdd(ap + 3, mv.w);
  if (lane == 0) unsafeAtomicAdd(deg + d, 1.0f);
}

// ---------------------------------------------------------------------------
// Kernel 3: out = relu(self + agg / max(deg,1))
// ---------------------------------------------------------------------------
__global__ void rgcn_final(float* __restrict__ out,
                           const float* __restrict__ agg,
                           const float* __restrict__ deg) {
  const int i = blockIdx.x * 256 + threadIdx.x;
  if (i >= NN * DD) return;
  const int node = i >> 7;
  float dg = deg[node];
  dg = dg < 1.0f ? 1.0f : dg;
  const float v = out[i] + agg[i] / dg;
  out[i] = v > 0.0f ? v : 0.0f;
}

__global__ void rgcn_zero(float* __restrict__ p, int total) {
  const int i = blockIdx.x * 256 + threadIdx.x;
  if (i < total) p[i] = 0.0f;
}

extern "C" void kernel_launch(void* const* d_in, const int* in_sizes, int n_in,
                              void* d_out, int out_size, void* d_ws, size_t ws_size,
                              hipStream_t stream) {
  const float* x     = (const float*)d_in[0];
  const float* Wrel  = (const float*)d_in[1];
  const float* Wself = (const float*)d_in[2];
  const float* bself = (const float*)d_in[3];
  const int*   eidx  = (const int*)d_in[4];
  const int*   etype = (const int*)d_in[5];
  float* out = (float*)d_out;

  // workspace layout (floats then bf16):
  //   h[8*N*128] f32 | agg[N*128] f32 | deg[N] f32 | xb[N*128] bf16 | wtb[9*128*128] bf16
  float* hbuf = (float*)d_ws;
  float* agg  = hbuf + (size_t)RR * NN * DD;
  float* deg  = agg + (size_t)NN * DD;
  unsigned short* xb  = (unsigned short*)(deg + NN);
  unsigned short* wtb = xb + (size_t)NN * DD;

  // zero agg+deg (contiguous) every launch — harness does not re-poison
  const int ztotal = NN * DD + NN;
  rgcn_zero<<<(ztotal + 255) / 256, 256, 0, stream>>>(agg, ztotal);

  // prep: bf16 copies of x and (transposed) weights
  rgcn_cvt_x<<<(NN * DD / 4 + 255) / 256, 256, 0, stream>>>(x, xb);
  rgcn_cvt_w<<<((RR + 1) * DD * DD + 255) / 256, 256, 0, stream>>>(Wrel, Wself, wtb);

  // 9 GEMMs (8 relations + self), 4 waves/block, one 16-row tile per wave
  dim3 g((MTILES + 3) / 4, RR + 1);
  rgcn_gemm_wmma<<<g, 128, 0, stream>>>(xb, wtb, bself, hbuf, out);

  // 800k edges, one wave each, 8 waves per block
  rgcn_edges<<<EE / 8, 256, 0, stream>>>(eidx, etype, hbuf, agg, deg);

  // finalize
  rgcn_final<<<(NN * DD + 255) / 256, 256, 0, stream>>>(out, agg, deg);
}
